// LinearFLHGS_1176821039699
// MI455X (gfx1250) — 8-bit matrix operand order repaired and checked by reading, compiled, not run on hardware
//
#include <hip/hip_runtime.h>
#include <hip/hip_bf16.h>

#define B_DIM 4096
#define O_DIM 11008
#define K_DIM 4096
#define GS_   128
#define G_DIM (K_DIM / GS_)

typedef __attribute__((ext_vector_type(16))) int   v16i;
typedef __attribute__((ext_vector_type(8)))  float v8f;

typedef int vint4_t __attribute__((__vector_size__(4 * sizeof(int))));
typedef __attribute__((address_space(1))) vint4_t glb_vint4;
typedef __attribute__((address_space(3))) vint4_t lds_vint4;

#if defined(__has_builtin)
#  if __has_builtin(__builtin_amdgcn_global_load_async_to_lds_b128)
#    define HAVE_ASYNC_LDS 1
#  endif
#endif

#if defined(HAVE_ASYNC_LDS) && __has_builtin(__builtin_amdgcn_s_wait_asynccnt)
#  define WAIT_ASYNC(n) __builtin_amdgcn_s_wait_asynccnt(n)
#elif defined(HAVE_ASYNC_LDS)
#  define WAIT_ASYNC(n) asm volatile("s_wait_asynccnt %0" :: "i"(n) : "memory")
#else
#  define WAIT_ASYNC(n) ((void)0)
#endif

__device__ __forceinline__ unsigned int nib2fp8(int nib) {
    const unsigned long long lut_lo = 0x4E4C4A4844403800ULL;
    const unsigned long long lut_hi = 0xB8C0C4C8CACCCED0ULL;
    unsigned long long l = (nib & 8) ? lut_hi : lut_lo;
    return (unsigned int)(l >> ((nib & 7) * 8)) & 0xffu;
}

__global__ __launch_bounds__(256)
void repack_int4_fp8_kernel(const int* __restrict__ packed,
                            unsigned int* __restrict__ out_fp8_words,
                            int n_packed)
{
    int i = blockIdx.x * blockDim.x + threadIdx.x;
    int base = i * 2;
    if (base + 1 >= n_packed) return;
    int2 p = *(const int2*)(packed + base);
    unsigned int b0 = nib2fp8( p.x        & 15);
    unsigned int b1 = nib2fp8((p.x >> 4)  & 15);
    unsigned int b2 = nib2fp8( p.y        & 15);
    unsigned int b3 = nib2fp8((p.y >> 4)  & 15);
    out_fp8_words[i] = b0 | (b1 << 8) | (b2 << 16) | (b3 << 24);
}

__device__ __forceinline__ int sw_off(int row, int byte_off) {
    int chunk  = byte_off >> 4;
    int within = byte_off & 15;
    return row * 128 + (((chunk ^ (row & 7)) << 4) | within);
}

__device__ __forceinline__
void stage_tile(const unsigned char* __restrict__ gsrc_row, unsigned char* lbase,
                int ld_row, int ld_off)
{
#if defined(HAVE_ASYNC_LDS)
#pragma unroll
    for (int q = 0; q < 4; ++q) {
        const int boff = ld_off + q * 16;
        __builtin_amdgcn_global_load_async_to_lds_b128(
            (glb_vint4*)(gsrc_row + boff),
            (lds_vint4*)(lbase + sw_off(ld_row, boff)),
            0, 0);
    }
#else
    int4 r[4];
#pragma unroll
    for (int q = 0; q < 4; ++q)
        r[q] = *(const int4*)(gsrc_row + ld_off + q * 16);
#pragma unroll
    for (int q = 0; q < 4; ++q)
        *(int4*)(lbase + sw_off(ld_row, ld_off + q * 16)) = r[q];
#endif
}

#define TILE_BYTES (128 * 128)

__global__ __launch_bounds__(256, 2)
void gemm_w4a4_kernel(const unsigned char* __restrict__ X8,
                      const unsigned char* __restrict__ W8,
                      const _Float16* __restrict__ a_scale,
                      const _Float16* __restrict__ w_scale,
                      const _Float16* __restrict__ bias,
                      _Float16* __restrict__ out)
{
    __shared__ __attribute__((aligned(16))) unsigned char lx[2][TILE_BYTES];
    __shared__ __attribute__((aligned(16))) unsigned char lw[2][TILE_BYTES];

    const int tid  = threadIdx.x;
    const int wave = tid >> 5;
    const int lane = tid & 31;
    const int half = lane >> 4;
    const int l15  = lane & 15;

    const int m0 = blockIdx.y * 128;
    const int n0 = blockIdx.x * 128;

    v8f accf[8] = {};

    const int ld_row = tid >> 1;
    const int ld_off = (tid & 1) * 64;
    const unsigned char* xg = X8 + (size_t)(m0 + ld_row) * K_DIM;
    const unsigned char* wg = W8 + (size_t)(n0 + ld_row) * K_DIM;

    stage_tile(xg, lx[0], ld_row, ld_off);
    stage_tile(wg, lw[0], ld_row, ld_off);

    int buf = 0;
    for (int g = 0; g < G_DIM; ++g) {
        if (g + 1 < G_DIM) {
            stage_tile(xg + (g + 1) * GS_, lx[buf ^ 1], ld_row, ld_off);
            stage_tile(wg + (g + 1) * GS_, lw[buf ^ 1], ld_row, ld_off);
            WAIT_ASYNC(8);
        } else {
            WAIT_ASYNC(0);
        }
        __syncthreads();

        const int ra = wave * 16 + l15;
        const unsigned char* xrow = lx[buf] + ra * 128;
        const int rx = ra & 7;
        v16i A;
#pragma unroll
        for (int p = 0; p < 4; ++p) {
            int4 d0 = *(const int4*)(xrow + (((2 * p + half) ^ rx) << 4));
            A[4 * p]     = d0.x; A[4 * p + 1] = d0.y;
            A[4 * p + 2] = d0.z; A[4 * p + 3] = d0.w;
        }

        float a_s[8];
#pragma unroll
        for (int r = 0; r < 8; ++r)
            a_s[r] = (float)a_scale[(size_t)(m0 + wave * 16 + half * 8 + r) * G_DIM + g];

#pragma unroll
        for (int nt = 0; nt < 8; ++nt) {
            const int rb = nt * 16 + l15;
            const unsigned char* wrow = lw[buf] + rb * 128;
            const int rw = rb & 7;
            int4 q0 = *(const int4*)(wrow + (((half    ) ^ rw) << 4));
            int4 q1 = *(const int4*)(wrow + (((2 + half) ^ rw) << 4));
            int4 q2 = *(const int4*)(wrow + (((4 + half) ^ rw) << 4));
            int4 q3 = *(const int4*)(wrow + (((6 + half) ^ rw) << 4));
            v16i Bv;
            Bv[0]  = q0.x; Bv[1]  = q0.y; Bv[2]  = q0.z; Bv[3]  = q0.w;
            Bv[4]  = q1.x; Bv[5]  = q1.y; Bv[6]  = q1.z; Bv[7]  = q1.w;
            Bv[8]  = q2.x; Bv[9]  = q2.y; Bv[10] = q2.z; Bv[11] = q2.w;
            Bv[12] = q3.x; Bv[13] = q3.y; Bv[14] = q3.z; Bv[15] = q3.w;

            v8f cz = {};
            v8f d = __builtin_amdgcn_wmma_f32_16x16x128_fp8_fp8(
                        A, Bv, (short)0, cz, false, false);

            float w_s = (float)w_scale[(size_t)(n0 + nt * 16 + l15) * G_DIM + g];
#pragma unroll
            for (int r = 0; r < 8; ++r)
                accf[nt][r] += d[r] * (a_s[r] * w_s);
        }
        __syncthreads();
        buf ^= 1;
    }

#pragma unroll
    for (int nt = 0; nt < 8; ++nt) {
        const int o = n0 + nt * 16 + l15;
        const float bv = (float)bias[o];
#pragma unroll
        for (int r = 0; r < 8; ++r) {
            const int m = m0 + wave * 16 + half * 8 + r;
            out[(size_t)m * O_DIM + o] = (_Float16)(accf[nt][r] + bv);
        }
    }
}

extern "C" void kernel_launch(void* const* d_in, const int* in_sizes, int n_in,
                              void* d_out, int out_size, void* d_ws, size_t ws_size,
                              hipStream_t stream)
{
    const int*      x        = (const int*)d_in[0];
    const _Float16* a_scale  = (const _Float16*)d_in[1];
    const int*      w_packed = (const int*)d_in[2];
    const _Float16* w_scale  = (const _Float16*)d_in[3];
    const _Float16* bias     = (const _Float16*)d_in[4];
    _Float16*       out      = (_Float16*)d_out;

    unsigned char* X8 = (unsigned char*)d_ws;
    unsigned char* W8 = X8 + (size_t)B_DIM * K_DIM;

    {
        int n_packed  = B_DIM * (K_DIM / 2);
        int n_threads = n_packed / 2;
        repack_int4_fp8_kernel<<<n_threads / 256, 256, 0, stream>>>(
            x, (unsigned int*)X8, n_packed);
    }
    {
        int n_packed  = O_DIM * (K_DIM / 2);
        int n_threads = n_packed / 2;
        repack_int4_fp8_kernel<<<n_threads / 256, 256, 0, stream>>>(
            w_packed, (unsigned int*)W8, n_packed);
    }

    dim3 grid(O_DIM / 128, B_DIM / 128);
    gemm_w4a4_kernel<<<grid, 256, 0, stream>>>(X8, W8, a_scale, w_scale, bias, out);
}
